// TopwCrossAttention_35107062677738
// MI455X (gfx1250) — matrix operand order repaired and checked by reading, not compiled, not run on hardware
//
#include <hip/hip_runtime.h>

// ---------------------------------------------------------------- constants
#define B_     4
#define NQ_    256
#define DQ_    256
#define DKV_   256
#define HIMG   256
#define WIMG   256
#define HEADS_ 8
#define DH_    32
#define TOPW_  4
#define WS_    8
#define INNER_ 256
#define DOUT_  256
#define NWIN   256   // 16x16 windows per image
#define TOK    64    // tokens per window
#define NKEY   256   // TOPW * TOK keys per query

typedef __attribute__((ext_vector_type(16))) _Float16 v16h;
typedef __attribute__((ext_vector_type(8)))  float    v8f;
typedef __attribute__((ext_vector_type(4)))  unsigned int u32x4;
typedef __attribute__((ext_vector_type(8)))  int      i32x8;
typedef __attribute__((ext_vector_type(4)))  int      i32x4;

#if defined(__AMDGCN__)
#define HAS_TDM        __has_builtin(__builtin_amdgcn_tensor_load_to_lds)
#define HAS_ASYNC_LDS  __has_builtin(__builtin_amdgcn_global_load_async_to_lds_b128)
#define HAS_WAIT_ASYNC __has_builtin(__builtin_amdgcn_s_wait_asynccnt)
#define HAS_WAIT_TENS  __has_builtin(__builtin_amdgcn_s_wait_tensorcnt)
#else
#define HAS_TDM 0
#define HAS_ASYNC_LDS 0
#define HAS_WAIT_ASYNC 0
#define HAS_WAIT_TENS 0
#endif

// Compile-time probes (device pass only): confirm data-mover availability.
#if defined(__AMDGCN__)
#if HAS_TDM
#warning "CDNA5-probe: tensor_load_to_lds AVAILABLE (TDM path active)"
#else
#warning "CDNA5-probe: tensor_load_to_lds NOT available"
#endif
#if HAS_ASYNC_LDS
#warning "CDNA5-probe: global_load_async_to_lds_b128 AVAILABLE (async path active)"
#else
#warning "CDNA5-probe: global_load_async_to_lds_b128 NOT available"
#endif
#endif

// ---------------------------------------------------------------- async copy
// 16-byte global->LDS copy: async DMA (ASYNCcnt) when available, else via VGPR.
// Builtin signature (from hipcc diagnostic): arg0 = int4 addrspace(1)* (global),
// arg1 = int4 addrspace(3)* (LDS), then imm offset, imm cpol.
typedef int v4i_vs __attribute__((vector_size(16)));
#if HAS_ASYNC_LDS
typedef __attribute__((address_space(1))) v4i_vs gv4_t;
typedef __attribute__((address_space(3))) v4i_vs lv4_t;
#endif

__device__ __forceinline__ void async_copy16(const void* gsrc, void* ldst) {
#if HAS_ASYNC_LDS
  __builtin_amdgcn_global_load_async_to_lds_b128(
      (gv4_t*)(unsigned long long)gsrc,
      (lv4_t*)(unsigned int)(unsigned long long)ldst, 0, 0);
#else
  *(uint4*)ldst = *(const uint4*)gsrc;
#endif
}
__device__ __forceinline__ void async_join() {
#if HAS_ASYNC_LDS && HAS_WAIT_ASYNC
  __builtin_amdgcn_s_wait_asynccnt(0);
#endif
}

// ---------------------------------------------------------------- WMMA frags
// A: 16x32 f16 tile, row-major with leading dim lda.
// lane<16 -> K 0..15, lane>=16 -> K 16..31; elem e -> K=base+e.  Row = lane&15.
__device__ __forceinline__ v16h load_a16(const _Float16* A, int lda, int lane) {
  int row  = lane & 15;
  int base = (lane >> 4) << 4;
  const _Float16* p0 = A + (long)row * lda + base;
  v16h f;
#pragma unroll
  for (int e = 0; e < 16; ++e) f[e] = p0[e];
  return f;
}

// B: 32x16 f16 tile given as B^T row-major (N rows x K cols, leading dim ldb).
// Lane n=lane&15 -> column n; lanes 16..31 carry K=16..31.
__device__ __forceinline__ v16h load_bt16(const _Float16* Bt, int ldb, int lane) {
  int n  = lane & 15;
  int kb = (lane >> 4) << 4;
  const _Float16* p = Bt + (long)n * ldb + kb;
  v16h f;
#pragma unroll
  for (int e = 0; e < 16; ++e) f[e] = p[e];
  return f;
}

__device__ __forceinline__ v8f wmma16(v16h a, v16h b, v8f c) {
  return __builtin_amdgcn_wmma_f32_16x16x32_f16(false, a, false, b, (short)0, c,
                                                false, false);
}

// ---------------------------------------------------------------- reductions
__device__ __forceinline__ float block_reduce_sum(float v, float* red) {
#pragma unroll
  for (int o = 16; o > 0; o >>= 1) v += __shfl_xor(v, o, 32);
  __syncthreads();
  if ((threadIdx.x & 31) == 0) red[threadIdx.x >> 5] = v;
  __syncthreads();
  float s = 0.f;
#pragma unroll
  for (int i = 0; i < 8; ++i) s += red[i];
  return s;
}

__device__ __forceinline__ float block_reduce_max(float v, float* red) {
#pragma unroll
  for (int o = 16; o > 0; o >>= 1) v = fmaxf(v, __shfl_xor(v, o, 32));
  __syncthreads();
  if ((threadIdx.x & 31) == 0) red[threadIdx.x >> 5] = v;
  __syncthreads();
  float s = red[0];
#pragma unroll
  for (int i = 1; i < 8; ++i) s = fmaxf(s, red[i]);
  return s;
}

// ---------------------------------------------------------------- f32 -> f16
__global__ void cvt_f32_f16_kernel(const float* __restrict__ s,
                                   _Float16* __restrict__ d, int n) {
  int i = blockIdx.x * blockDim.x + threadIdx.x;
  if (i < n) d[i] = (_Float16)s[i];
}

// ---------------------------------------------------------------- conv GEMM
// imagec = Conv2d(image, k=2, s=2) written directly in window/token layout:
// xw[b][win][tok][c] (f16).  Implicit GEMM: M=65536 patch rows, N=256, K=1024.
__global__ __launch_bounds__(256)
void conv_patchify_kernel(const float* __restrict__ image,
                          const _Float16* __restrict__ cw16,  // 256 x 1024 (N x K)
                          const float* __restrict__ conv_b,
                          _Float16* __restrict__ xw) {
  __shared__ _Float16 Ash[128][64];   // 16 KB: 128 rows x 64-K chunk
  __shared__ _Float16 Bsh[256][64];   // 32 KB: 256 N x 64-K chunk of weights
  const int tid  = threadIdx.x;
  const int lane = tid & 31;
  const int wave = tid >> 5;
  const int r0   = blockIdx.x * 128;

  v8f acc[16] = {};

  for (int kc = 0; kc < 16; ++kc) {          // 16 chunks of 64 K (= 16 channels)
    __syncthreads();
    // --- async-stage the weight chunk (DMA overlaps the A gather below) ---
#pragma unroll
    for (int i = 0; i < 8; ++i) {            // 256 rows x 8 x 16B = 2048 segs
      int task = tid + i * 256;
      int row = task >> 3;
      int seg = task & 7;
      async_copy16(cw16 + (long)row * 1024 + kc * 64 + seg * 8,
                   &Bsh[row][seg * 8]);
    }
    // --- gather + convert A patch rows ---
#pragma unroll
    for (int it = 0; it < 8; ++it) {         // 128 rows x 16 ch = 2048 tasks
      int task = tid + it * 256;
      int rl = task >> 4;
      int ch = task & 15;
      int r  = r0 + rl;
      int bb = r >> 14;
      int oy = (r >> 7) & 127;
      int ox = r & 127;
      int i  = kc * 16 + ch;
      const float* src =
          image + (((long)(bb * DKV_ + i) * HIMG) + 2 * oy) * WIMG + 2 * ox;
      float v00 = src[0],     v01 = src[1];
      float v10 = src[WIMG],  v11 = src[WIMG + 1];
      if (kc < 15) {  // pull next chunk's lines toward L2 (global_prefetch_b8)
        __builtin_prefetch(src + (long)16 * HIMG * WIMG, 0, 1);
      }
      _Float16* d = &Ash[rl][ch * 4];
      d[0] = (_Float16)v00; d[1] = (_Float16)v01;
      d[2] = (_Float16)v10; d[3] = (_Float16)v11;
    }
    async_join();
    __syncthreads();
#pragma unroll
    for (int ks = 0; ks < 2; ++ks) {
      v16h a = load_a16(&Ash[wave * 16][ks * 32], 64, lane);
#pragma unroll
      for (int nt = 0; nt < 16; ++nt) {
        v16h b = load_bt16(&Bsh[nt * 16][ks * 32], 64, lane);
        acc[nt] = wmma16(a, b, acc[nt]);
      }
    }
  }

  // epilogue: + bias, scatter into dilated-window/token layout
  const int colb = lane & 15;
  const int rb   = (lane >> 4) << 3;
#pragma unroll
  for (int nt = 0; nt < 16; ++nt) {
#pragma unroll
    for (int r = 0; r < 8; ++r) {
      int gr = r0 + wave * 16 + rb + r;
      int bb = gr >> 14;
      int oy = (gr >> 7) & 127;
      int ox = gr & 127;
      int m = oy & 15, wsh = oy >> 4;
      int n = ox & 15, wsw = ox >> 4;
      int win = m * 16 + n;
      int t   = wsh * 8 + wsw;
      int c   = nt * 16 + colb;
      float v = acc[nt][r] + conv_b[c];
      xw[(((long)(bb * NWIN + win)) * TOK + t) * INNER_ + c] = (_Float16)v;
    }
  }
}

// ---------------------------------------------------------------- window pool
// Per window: cor = x x^T * DKV^-0.5 ; softmax_one rows; col-sum; softmax_one;
// imagew[c] = sum_q x[q][c] * w[q]
__global__ __launch_bounds__(256)
void window_pool_kernel(const _Float16* __restrict__ xw,
                        _Float16* __restrict__ imagew16) {
  __shared__ _Float16 xwin[TOK][INNER_];  // 32 KB
  __shared__ float cor[TOK][TOK];         // 16 KB
  __shared__ float sline[TOK];
  __shared__ float wline[TOK];
  const int tid  = threadIdx.x;
  const int lane = tid & 31;
  const int wave = tid >> 5;
  const long wbase = (long)blockIdx.x * TOK * INNER_;

  // ---- stage the 32 KB window tile into LDS ----
#if HAS_TDM
  if (wave == 0) {
    // Tensor DMA: 2-byte elements, 1 row of 16384 elems, contiguous.
    unsigned int lds_base = (unsigned int)(unsigned long long)(&xwin[0][0]);
    unsigned long long ga = (unsigned long long)(xw + wbase);
    u32x4 g0;
    g0[0] = 1u;                                  // count=1, load, no gather
    g0[1] = lds_base;                            // lds_addr (bytes)
    g0[2] = (unsigned int)ga;                    // global_addr[31:0]
    g0[3] = (unsigned int)((ga >> 32) & 0x01FFFFFFull) | 0x80000000u;  // type=2
    i32x8 g1;
    g1[0] = 0x00010000;   // data_size=1 (2 bytes)
    g1[1] = 0x40000000;   // tensor_dim0 = 16384 (low16 in [31:16])
    g1[2] = 0x00010000;   // tensor_dim0 hi=0 | tensor_dim1 = 1
    g1[3] = 0x40000000;   // tensor_dim1 hi=0 | tile_dim0 = 16384
    g1[4] = 0x00000001;   // tile_dim1 = 1, tile_dim2 = 0
    g1[5] = 16384;        // tensor_dim0_stride low32
    g1[6] = 0;            // stride0 hi | stride1 lo
    g1[7] = 0;            // stride1 hi
    i32x4 gz; gz[0] = 0; gz[1] = 0; gz[2] = 0; gz[3] = 0;
#if __clang_major__ >= 23
    i32x8 gz8; for (int i = 0; i < 8; ++i) gz8[i] = 0;
    __builtin_amdgcn_tensor_load_to_lds(g0, g1, gz, gz, gz8, 0);
#else
    __builtin_amdgcn_tensor_load_to_lds(g0, g1, gz, gz, 0);
#endif
#if HAS_WAIT_TENS
    __builtin_amdgcn_s_wait_tensorcnt(0);
#endif
  }
#else
  {
#pragma unroll
    for (int i = 0; i < 8; ++i) {
      int o = (tid + i * 256) * 16;
      async_copy16((const char*)(xw + wbase) + o, (char*)(&xwin[0][0]) + o);
    }
    async_join();
  }
#endif
  __syncthreads();

  // cor GEMM: 4x4 tiles of 16x16, K=256. 8 waves x 2 tiles.
#pragma unroll
  for (int j = 0; j < 2; ++j) {
    int tile = wave * 2 + j;
    int pt = tile >> 2, qt = tile & 3;
    v8f acc = {};
    for (int k0 = 0; k0 < INNER_; k0 += 32) {
      v16h a = load_a16(&xwin[pt * 16][k0], INNER_, lane);
      v16h b = load_bt16(&xwin[qt * 16][k0], INNER_, lane);  // B^T == x
      acc = wmma16(a, b, acc);
    }
    int col = lane & 15, rb = (lane >> 4) << 3;
#pragma unroll
    for (int r = 0; r < 8; ++r)
      cor[pt * 16 + rb + r][qt * 16 + col] = acc[r] * 0.0625f;  // DKV^-0.5
  }
  __syncthreads();

  if (tid < TOK) {  // softmax_one along rows
    int p = tid;
    float m = -1e30f;
    for (int q = 0; q < TOK; ++q) m = fmaxf(m, cor[p][q]);
    float s = 0.f;
    for (int q = 0; q < TOK; ++q) { float e = expf(cor[p][q] - m); cor[p][q] = e; s += e; }
    float inv = 1.f / (1.f + s);
    for (int q = 0; q < TOK; ++q) cor[p][q] *= inv;
  }
  __syncthreads();
  if (tid < TOK) {  // sum over rows (axis=1)
    int q = tid;
    float s = 0.f;
    for (int p = 0; p < TOK; ++p) s += cor[p][q];
    sline[q] = s;
  }
  __syncthreads();
  if (tid < TOK) {  // softmax_one over the 64 column sums
    float m = -1e30f;
    for (int q = 0; q < TOK; ++q) m = fmaxf(m, sline[q]);
    float s = 0.f;
    for (int q = 0; q < TOK; ++q) s += expf(sline[q] - m);
    wline[tid] = expf(sline[tid] - m) / (1.f + s);
  }
  __syncthreads();
  {  // weighted token sum -> window summary
    int c = tid;
    float a = 0.f;
    for (int q = 0; q < TOK; ++q) a += (float)xwin[q][c] * wline[q];
    imagew16[(long)blockIdx.x * INNER_ + c] = (_Float16)a;
  }
}

// ---------------------------------------------------------------- layernorm
__global__ __launch_bounds__(256)
void layernorm_kernel(const float* __restrict__ src,
                      const float* __restrict__ g,
                      const float* __restrict__ bb,
                      _Float16* __restrict__ out16,
                      float* __restrict__ out32) {
  __shared__ float red[8];
  int row = blockIdx.x, c = threadIdx.x;
  float v  = src[(long)row * 256 + c];
  float mu = block_reduce_sum(v, red) * (1.f / 256.f);
  float d  = v - mu;
  float var = block_reduce_sum(d * d, red) * (1.f / 256.f);
  float y = d * rsqrtf(var + 1e-5f) * g[c] + bb[c];
  if (out16) out16[(long)row * 256 + c] = (_Float16)y;
  if (out32) out32[(long)row * 256 + c] = y;
}

// ---------------------------------------------------------------- generic GEMM
// C[MxN] = A[MxK] * B^T ; N=K=256 fixed. 128 rows per block, 8 waves.
// B K-chunk async-staged in LDS once per block (kills 8x redundant loads).
__global__ __launch_bounds__(256)
void gemm128_kernel(const _Float16* __restrict__ A, long sA,
                    const _Float16* __restrict__ Bt, long sB,
                    const float* __restrict__ bias,
                    const float* __restrict__ resid, long sR,
                    float* __restrict__ out32, long sO,
                    _Float16* __restrict__ out16, long sO16,
                    int relu) {
  __shared__ _Float16 Bsh[256][32];   // 16 KB: 256 N x 32-K chunk
  const int tid = threadIdx.x, lane = tid & 31, wave = tid >> 5;
  const int z  = blockIdx.z;
  const int r0 = blockIdx.x * 128 + wave * 16;
  const _Float16* Ab = A  + (long)z * sA;
  const _Float16* Bb = Bt + (long)z * sB;

  v8f acc[16] = {};
  for (int k0 = 0; k0 < 256; k0 += 32) {
    __syncthreads();
#pragma unroll
    for (int i = 0; i < 4; ++i) {    // 256 rows x 4 x 16B = 1024 segs
      int task = tid + i * 256;
      int row = task >> 2;
      int seg = task & 3;
      async_copy16(Bb + (long)row * 256 + k0 + seg * 8, &Bsh[row][seg * 8]);
    }
    async_join();
    __syncthreads();
    v16h a = load_a16(Ab + (long)r0 * 256 + k0, 256, lane);
#pragma unroll
    for (int nt = 0; nt < 16; ++nt) {
      v16h b = load_bt16(&Bsh[nt * 16][0], 32, lane);
      acc[nt] = wmma16(a, b, acc[nt]);
    }
  }
  const int col0 = lane & 15, rb = (lane >> 4) << 3;
#pragma unroll
  for (int nt = 0; nt < 16; ++nt) {
    int c = nt * 16 + col0;
    float bv = bias ? bias[c] : 0.f;
#pragma unroll
    for (int r = 0; r < 8; ++r) {
      int gr = r0 + rb + r;
      long idx = (long)gr * 256 + c;
      float v = acc[nt][r] + bv;
      if (resid) v += resid[(long)z * sR + idx];
      if (relu)  v = fmaxf(v, 0.f);
      if (out32) out32[(long)z * sO + idx] = v;
      if (out16) out16[(long)z * sO16 + idx] = (_Float16)v;
    }
  }
}

// ---------------------------------------------------------------- top-4 routing
__global__ void topk4_kernel(const float* __restrict__ rattn,
                             int* __restrict__ topidx) {
  int row = blockIdx.x * blockDim.x + threadIdx.x;
  if (row >= B_ * NQ_) return;
  const float* rp = rattn + (long)row * NWIN;
  float bv[4] = {-1e30f, -1e30f, -1e30f, -1e30f};
  int   bi[4] = {0, 0, 0, 0};
  for (int m = 0; m < NWIN; ++m) {
    float v = rp[m];
    for (int j = 0; j < 4; ++j) {
      if (v > bv[j]) {
        for (int k = 3; k > j; --k) { bv[k] = bv[k - 1]; bi[k] = bi[k - 1]; }
        bv[j] = v; bi[j] = m;
        break;
      }
    }
  }
  for (int j = 0; j < 4; ++j) topidx[row * 4 + j] = bi[j];
}

// ---------------------------------------------------------------- fused attn
// One block per query: gather 4 windows, LN (stats precomputed once), per-head
// K/V projection via WMMA from a 32-channel LDS-staged kv chunk, softmax, AV.
__global__ __launch_bounds__(256)
void topw_attention_kernel(const _Float16* __restrict__ xw,
                           const float* __restrict__ pe,
                           const float* __restrict__ gkv,
                           const float* __restrict__ bkv,
                           const _Float16* __restrict__ wk16,
                           const _Float16* __restrict__ wv16,
                           const float* __restrict__ q32,
                           const int* __restrict__ topidx,
                           _Float16* __restrict__ attnout16) {
  __shared__ _Float16 kvch[NKEY][32];   // 16 KB
  __shared__ _Float16 kh[NKEY][32];     // 16 KB
  __shared__ _Float16 vh[NKEY][32];     // 16 KB
  __shared__ float mu[NKEY], rsig[NKEY];
  __shared__ float qrow[INNER_];
  __shared__ float gg[INNER_], bbv[INNER_];
  __shared__ float attnv[NKEY];
  __shared__ float pscr[256];
  __shared__ float red[8];
  __shared__ int widx[TOPW_];

  const int tid = threadIdx.x, lane = tid & 31, wave = tid >> 5;
  const int g  = blockIdx.x;            // b*NQ + qi
  const int bb = g >> 8;

  if (tid < TOPW_) widx[tid] = topidx[g * TOPW_ + tid];
  qrow[tid] = q32[(long)g * INNER_ + tid];
  gg[tid]  = gkv[tid];
  bbv[tid] = bkv[tid];
  __syncthreads();

  const int t = tid;  // one token per thread
  const _Float16* xrow =
      xw + (((long)(bb * NWIN + widx[t >> 6])) * TOK + (t & 63)) * INNER_;
  {  // LN stats over 256 channels of (x + pos_embed)
    float s = 0.f, ss = 0.f;
    for (int c = 0; c < INNER_; ++c) {
      float v = (float)xrow[c] + pe[(long)t * INNER_ + c];
      s += v; ss += v * v;
    }
    float m = s * (1.f / INNER_);
    float var = ss * (1.f / INNER_) - m * m;
    mu[t]   = m;
    rsig[t] = rsqrtf(var + 1e-5f);
  }
  const float dscale = rsqrtf((float)DH_);

  for (int h = 0; h < HEADS_; ++h) {
    v8f accK[2][2] = {};
    v8f accV[2][2] = {};
    for (int kc = 0; kc < 8; ++kc) {    // stream kv in 32-channel chunks
      __syncthreads();
      {
        int c0 = kc * 32;
        float m = mu[t], rs = rsig[t];
#pragma unroll
        for (int j = 0; j < 32; ++j) {
          int c = c0 + j;
          float v = ((float)xrow[c] + pe[(long)t * INNER_ + c] - m) * rs * gg[c] + bbv[c];
          kvch[t][j] = (_Float16)v;
        }
      }
      __syncthreads();
#pragma unroll
      for (int rt = 0; rt < 2; ++rt) {
        v16h a = load_a16(&kvch[wave * 32 + rt * 16][0], 32, lane);
#pragma unroll
        for (int ct = 0; ct < 2; ++ct) {
          v16h bk = load_bt16(wk16 + (long)(h * 32 + ct * 16) * INNER_ + kc * 32,
                              INNER_, lane);
          v16h bv = load_bt16(wv16 + (long)(h * 32 + ct * 16) * INNER_ + kc * 32,
                              INNER_, lane);
          accK[rt][ct] = wmma16(a, bk, accK[rt][ct]);
          accV[rt][ct] = wmma16(a, bv, accV[rt][ct]);
        }
      }
    }
    __syncthreads();
    {  // spill K_h / V_h tiles to LDS (f16)
      int col = lane & 15, rb2 = (lane >> 4) << 3;
#pragma unroll
      for (int rt = 0; rt < 2; ++rt)
#pragma unroll
        for (int ct = 0; ct < 2; ++ct)
#pragma unroll
          for (int r = 0; r < 8; ++r) {
            int tok = wave * 32 + rt * 16 + rb2 + r;
            kh[tok][ct * 16 + col] = (_Float16)accK[rt][ct][r];
            vh[tok][ct * 16 + col] = (_Float16)accV[rt][ct][r];
          }
    }
    __syncthreads();

    // dots + softmax over the 256 keys
    float d = 0.f;
#pragma unroll
    for (int j = 0; j < DH_; ++j) d += qrow[h * DH_ + j] * (float)kh[t][j];
    d *= dscale;
    float mx = block_reduce_max(d, red);
    float e  = expf(d - mx);
    float s  = block_reduce_sum(e, red);
    attnv[t] = e / s;
    __syncthreads();

    {  // out_h[d] = sum_t attn[t] * V_h[t][d]
      int dd = tid & 31, grp = tid >> 5;
      float p = 0.f;
      for (int i = 0; i < 32; ++i) {
        int tok = grp + 8 * i;
        p += attnv[tok] * (float)vh[tok][dd];
      }
      pscr[grp * 32 + dd] = p;
    }
    __syncthreads();
    if (tid < 32) {
      float o = 0.f;
#pragma unroll
      for (int gi = 0; gi < 8; ++gi) o += pscr[gi * 32 + tid];
      attnout16[(long)g * INNER_ + h * DH_ + tid] = (_Float16)o;
    }
    __syncthreads();
  }
}

// ---------------------------------------------------------------- launcher
extern "C" void kernel_launch(void* const* d_in, const int* in_sizes, int n_in,
                              void* d_out, int out_size, void* d_ws, size_t ws_size,
                              hipStream_t stream) {
  const float* query    = (const float*)d_in[0];
  const float* image    = (const float*)d_in[1];
  const float* conv_w   = (const float*)d_in[2];
  const float* conv_b   = (const float*)d_in[3];
  const float* pe       = (const float*)d_in[4];
  const float* normq_g  = (const float*)d_in[5];
  const float* normq_b  = (const float*)d_in[6];
  const float* normkv_g = (const float*)d_in[7];
  const float* normkv_b = (const float*)d_in[8];
  const float* wq       = (const float*)d_in[9];
  const float* wk       = (const float*)d_in[10];
  const float* wv       = (const float*)d_in[11];
  const float* w_rec    = (const float*)d_in[12];
  const float* mlp_g    = (const float*)d_in[13];
  const float* mlp_b    = (const float*)d_in[14];
  const float* mlp_w1   = (const float*)d_in[15];
  const float* mlp_b1   = (const float*)d_in[16];
  const float* mlp_w2   = (const float*)d_in[17];
  const float* mlp_b2   = (const float*)d_in[18];

  char* ws = (char*)d_ws;
  size_t off = 0;
  auto alloc = [&](size_t bytes) -> void* {
    void* p = ws + off;
    off = (off + bytes + 255) & ~(size_t)255;
    return p;
  };

  _Float16* cw16     = (_Float16*)alloc((size_t)262144 * 2);  // 256x1024
  _Float16* wq16     = (_Float16*)alloc((size_t)65536 * 2);
  _Float16* wk16     = (_Float16*)alloc((size_t)65536 * 2);
  _Float16* wv16     = (_Float16*)alloc((size_t)65536 * 2);
  _Float16* wrec16   = (_Float16*)alloc((size_t)65536 * 2);
  _Float16* w1_16    = (_Float16*)alloc((size_t)65536 * 2);
  _Float16* w2_16    = (_Float16*)alloc((size_t)65536 * 2);
  _Float16* xw       = (_Float16*)alloc((size_t)16777216 * 2);  // 33.5 MB
  _Float16* imagew16 = (_Float16*)alloc((size_t)262144 * 2);
  _Float16* lnq16    = (_Float16*)alloc((size_t)262144 * 2);
  float*    q32      = (float*)alloc((size_t)262144 * 4);
  _Float16* q16      = (_Float16*)alloc((size_t)262144 * 2);
  float*    rattn    = (float*)alloc((size_t)262144 * 4);
  int*      topidx   = (int*)alloc((size_t)4096 * 4);
  _Float16* attnout16= (_Float16*)alloc((size_t)262144 * 2);
  float*    outbuf   = (float*)alloc((size_t)262144 * 4);
  _Float16* lnout16  = (_Float16*)alloc((size_t)262144 * 2);
  _Float16* h1_16    = (_Float16*)alloc((size_t)262144 * 2);

  // 0) weights -> f16 (conv_w flat (o, i*4+dy*2+dx) already matches GEMM K order)
  cvt_f32_f16_kernel<<<1024, 256, 0, stream>>>(conv_w, cw16, 262144);
  cvt_f32_f16_kernel<<<256, 256, 0, stream>>>(wq, wq16, 65536);
  cvt_f32_f16_kernel<<<256, 256, 0, stream>>>(wk, wk16, 65536);
  cvt_f32_f16_kernel<<<256, 256, 0, stream>>>(wv, wv16, 65536);
  cvt_f32_f16_kernel<<<256, 256, 0, stream>>>(w_rec, wrec16, 65536);
  cvt_f32_f16_kernel<<<256, 256, 0, stream>>>(mlp_w1, w1_16, 65536);
  cvt_f32_f16_kernel<<<256, 256, 0, stream>>>(mlp_w2, w2_16, 65536);

  // 1) conv patchify -> windowed tokens (f16, L2-resident working set)
  conv_patchify_kernel<<<512, 256, 0, stream>>>(image, cw16, conv_b, xw);

  // 2) window pooling -> window summaries (TDM/async-staged LDS tile)
  window_pool_kernel<<<B_ * NWIN, 256, 0, stream>>>(xw, imagew16);

  // 3) q = LN(query) @ wq^T
  layernorm_kernel<<<B_ * NQ_, 256, 0, stream>>>(query, normq_g, normq_b, lnq16, nullptr);
  gemm128_kernel<<<dim3(8, 1, 1), 256, 0, stream>>>(
      lnq16, 0, wq16, 0, nullptr, nullptr, 0, q32, 0, q16, 0, 0);

  // 4) routing scores + top-4 (scale is monotonic -> omitted)
  gemm128_kernel<<<dim3(2, 1, 4), 256, 0, stream>>>(
      q16, 65536, imagew16, 65536, nullptr, nullptr, 0, rattn, 65536, nullptr, 0, 0);
  topk4_kernel<<<4, 256, 0, stream>>>(rattn, topidx);

  // 5) fused gather + LN + K/V projection + attention
  topw_attention_kernel<<<B_ * NQ_, 256, 0, stream>>>(
      xw, pe, normkv_g, normkv_b, wk16, wv16, q32, topidx, attnout16);

  // out = attnout @ w_rec^T + query
  gemm128_kernel<<<dim3(8, 1, 1), 256, 0, stream>>>(
      attnout16, 0, wrec16, 0, nullptr, query, 0, outbuf, 0, nullptr, 0, 0);

  // 6) residual MLP
  layernorm_kernel<<<B_ * NQ_, 256, 0, stream>>>(outbuf, mlp_g, mlp_b, lnout16, nullptr);
  gemm128_kernel<<<dim3(8, 1, 1), 256, 0, stream>>>(
      lnout16, 0, w1_16, 0, mlp_b1, nullptr, 0, nullptr, 0, h1_16, 0, 1);
  gemm128_kernel<<<dim3(8, 1, 1), 256, 0, stream>>>(
      h1_16, 0, w2_16, 0, mlp_b2, outbuf, 0, (float*)d_out, 0, nullptr, 0, 0);
}
